// attention_ST_A_embed_69801808494886
// MI455X (gfx1250) — matrix operand order repaired and checked by reading, not compiled, not run on hardware
//
#include <hip/hip_runtime.h>
#include <math.h>

typedef __attribute__((ext_vector_type(16))) _Float16 v16h;
typedef __attribute__((ext_vector_type(8)))  _Float16 v8h;
typedef __attribute__((ext_vector_type(8)))  float    v8f;
typedef __attribute__((ext_vector_type(4)))  float    f32x4;

#define N_   325
#define NP   336      // padded N (21 tiles of 16)
#define E_   64
#define NT   21       // row/col tiles
#define WSTR 352      // wT K-stride (11 chunks of 32)
#define NB   (NP/8)   // 8-col blocks per row (42)
#define NTHREADS 672  // 21 waves: one 16-row stripe per wave in both GEMM phases
#define NWAVES 21

union V16 { v16h h; f32x4 f[2]; };
union U8H { v8h h; uint4 u; };

__launch_bounds__(NTHREADS)
__global__ void attn_st_kernel(const float* __restrict__ X,
                               const float* __restrict__ Qw,
                               const float* __restrict__ Kw,
                               const float* __restrict__ adj,
                               const float* __restrict__ w,
                               float* __restrict__ out)
{
    // 225,792 B + 86,016 B = 311,808 B static LDS (<= 320 KB / workgroup)
    __shared__ _Float16 P[NP * NP];          // post-softmax probabilities (f16)
    __shared__ _Float16 QK[2 * NP * E_];     // phase1: q|k ; phase2 reuse: wT[64][352]

    const int tid  = threadIdx.x;
    const int lane = tid & 31;
    const int wv   = tid >> 5;
    const int hf   = lane >> 4;      // lane half (0/1)
    const int l15  = lane & 15;
    const int slice = blockIdx.x;
    const float* x = X + (size_t)slice * N_ * 3;

    // ---- build q, k (f16) in LDS; pad rows 325..335 with zeros ----
    for (int idx = tid; idx < NP * E_; idx += NTHREADS) {
        int n = idx >> 6, e = idx & 63;
        float qv = 0.f, kv = 0.f;
        if (n < N_) {
            float x0 = x[n*3+0], x1 = x[n*3+1], x2 = x[n*3+2];
            qv = x0*Qw[e] + x1*Qw[64+e] + x2*Qw[128+e];
            kv = x0*Kw[e] + x1*Kw[64+e] + x2*Kw[128+e];
        }
        QK[idx]         = (_Float16)qv;
        QK[NP*E_ + idx] = (_Float16)kv;
    }
    __syncthreads();

    // ---- phase 1: score = (q k^T)/8 * adj, mask zeros -> -inf, row softmax ----
    for (int i = wv; i < NT; i += NWAVES) {
        // A operand: q rows of tile i, K chunks 0..31 and 32..63
        const f32x4* qrow = (const f32x4*)&QK[(i*16 + l15) * E_];
        V16 a0, a1;
        a0.f[0] = qrow[hf];     a0.f[1] = qrow[2 + hf];
        a1.f[0] = qrow[4 + hf]; a1.f[1] = qrow[6 + hf];

        for (int j = 0; j < NT; ++j) {
            const f32x4* krow = (const f32x4*)&QK[NP*E_ + (j*16 + l15) * E_];
            V16 b0, b1;
            b0.f[0] = krow[hf];     b0.f[1] = krow[2 + hf];
            b1.f[0] = krow[4 + hf]; b1.f[1] = krow[6 + hf];

            v8f c = {0.f,0.f,0.f,0.f,0.f,0.f,0.f,0.f};
            c = __builtin_amdgcn_wmma_f32_16x16x32_f16(false, a0.h, false, b0.h,
                                                       (short)0, c, false, false);
            c = __builtin_amdgcn_wmma_f32_16x16x32_f16(false, a1.h, false, b1.h,
                                                       (short)0, c, false, false);
            int col  = j*16 + l15;
            int colc = (col < N_) ? col : (N_ - 1);
            bool colv = (col < N_);
            _Float16* pp = &P[(i*16 + 8*hf)*NP + col];   // rows fold to const offsets
            #pragma unroll
            for (int r = 0; r < 8; ++r) {
                int row  = i*16 + 8*hf + r;
                int rowc = (row < N_) ? row : (N_ - 1);
                float av = adj[rowc*N_ + colc];          // branchless, always in-bounds
                av = (colv && row < N_) ? av : 0.f;
                float s = c[r] * 0.125f * av;
                if (s == 0.f) s = -INFINITY;             // matches jnp.where(score==0,-inf)
                pp[r*NP] = (_Float16)s;
            }
        }
        asm volatile("s_wait_dscnt 0" ::: "memory");

        // softmax over the 16 rows of this stripe; lane pair (l, l+16) shares a row
        {
            int row = i*16 + l15;
            uint4* rowv4 = (uint4*)&P[row*NP];           // 42 x (8 f16) per row
            int b0 = hf*21, b1 = b0 + 21;

            U8H vmx;
            #pragma unroll
            for (int e = 0; e < 8; ++e) vmx.h[e] = (_Float16)(-INFINITY);
            for (int d = b0; d < b1; ++d) {
                U8H cv; cv.u = rowv4[d];
                vmx.h = __builtin_elementwise_max(vmx.h, cv.h);  // v_pk_max_num_f16
            }
            float m = -INFINITY;
            #pragma unroll
            for (int e = 0; e < 8; ++e) m = fmaxf(m, (float)vmx.h[e]);
            m = fmaxf(m, __shfl_xor(m, 16, 32));

            float sum = 0.f;
            if (m != -INFINITY) {
                for (int d = b0; d < b1; ++d) {
                    U8H cv; cv.u = rowv4[d];
                    #pragma unroll
                    for (int e = 0; e < 8; ++e) {
                        float ev = __expf((float)cv.h[e] - m);
                        sum += ev;
                        cv.h[e] = (_Float16)ev;
                    }
                    rowv4[d] = cv.u;
                }
            } else {
                uint4 z = {0u,0u,0u,0u};                 // fully masked / pad rows
                for (int d = b0; d < b1; ++d) rowv4[d] = z;
            }
            sum += __shfl_xor(sum, 16, 32);
            asm volatile("s_wait_dscnt 0" ::: "memory");
            if (m != -INFINITY) {
                _Float16 hinv = (_Float16)(1.f / sum);
                for (int d = b0; d < b1; ++d) {
                    U8H cv; cv.u = rowv4[d];
                    cv.h = cv.h * hinv;                  // v_pk_mul_f16
                    rowv4[d] = cv.u;
                }
            }
        }
    }
    __syncthreads();

    // ---- build wT[64][352] (zero K-pad >= 325) in the q/k region ----
    for (int idx = tid; idx < E_ * WSTR; idx += NTHREADS) {
        int e = idx / WSTR, kk = idx - e * WSTR;
        QK[idx] = (_Float16)((kk < N_) ? w[kk*E_ + e] : 0.f);
    }
    // ---- symmetrize P in place, 8-col blocks (race-free ownership) ----
    for (int idx = tid; idx < NP * NB; idx += NTHREADS) {
        int n  = idx / NB;
        int c0 = (idx - n * NB) * 8;
        if (c0 + 7 <= n) continue;                       // fully on/below diagonal
        if (c0 > n) {                                    // fully above diagonal
            U8H rv; rv.u = *(uint4*)&P[n*NP + c0];
            #pragma unroll
            for (int e = 0; e < 8; ++e) {
                int mm = c0 + e;
                float pb = (float)P[mm*NP + n];
                _Float16 avg = (_Float16)(((float)rv.h[e] + pb) * 0.5f);
                rv.h[e] = avg;
                P[mm*NP + n] = avg;
            }
            *(uint4*)&P[n*NP + c0] = rv.u;
        } else {                                         // diagonal-straddling block
            #pragma unroll
            for (int e = 0; e < 8; ++e) {
                int mm = c0 + e;
                if (mm > n) {
                    float pa = (float)P[n*NP + mm], pb = (float)P[mm*NP + n];
                    _Float16 avg = (_Float16)((pa + pb) * 0.5f);
                    P[n*NP + mm] = avg;
                    P[mm*NP + n] = avg;
                }
            }
        }
    }
    __syncthreads();

    // ---- phase 2: out = Psym @ w  (K = 352 as 11 chunks of 32) ----
    float* outs = out + (size_t)slice * N_ * E_;
    for (int i = wv; i < NT; i += NWAVES) {
        v8f acc[4];
        #pragma unroll
        for (int t = 0; t < 4; ++t)
            acc[t] = (v8f){0.f,0.f,0.f,0.f,0.f,0.f,0.f,0.f};

        const _Float16* prow = &P[(i*16 + l15) * NP];
        for (int cc = 0; cc < 11; ++cc) {
            int kb = cc * 32;
            V16 a;
            a.f[0] = *(const f32x4*)(prow + kb + hf*8);
            a.f[1] = *(const f32x4*)(prow + kb + 16 + hf*8);  // kb=320 spill hits zero-B rows
            #pragma unroll
            for (int t = 0; t < 4; ++t) {
                const f32x4* wr = (const f32x4*)&QK[(t*16 + l15) * WSTR + kb];
                V16 b; b.f[0] = wr[hf]; b.f[1] = wr[2 + hf];
                acc[t] = __builtin_amdgcn_wmma_f32_16x16x32_f16(false, a.h, false, b.h,
                                                               (short)0, acc[t], false, false);
            }
        }
        #pragma unroll
        for (int t = 0; t < 4; ++t) {
            #pragma unroll
            for (int r = 0; r < 8; ++r) {
                int row = i*16 + 8*hf + r;
                if (row < N_) outs[row*E_ + t*16 + l15] = acc[t][r];
            }
        }
    }
}

extern "C" void kernel_launch(void* const* d_in, const int* in_sizes, int n_in,
                              void* d_out, int out_size, void* d_ws, size_t ws_size,
                              hipStream_t stream) {
    const float* X  = (const float*)d_in[0];
    const float* Qw = (const float*)d_in[1];
    const float* Kw = (const float*)d_in[2];
    const float* A  = (const float*)d_in[3];
    const float* W  = (const float*)d_in[4];
    float* out = (float*)d_out;
    // one workgroup per (b,t) slice: 384 blocks x 672 threads (21 waves)
    attn_st_kernel<<<dim3(32*12), dim3(NTHREADS), 0, stream>>>(X, Qw, Kw, A, W, out);
}
